// AttentionHead_28638841929743
// MI455X (gfx1250) — compile-verified, matrix operand order checked by reading, not run on hardware
//
#include <hip/hip_runtime.h>
#include <cstddef>

// B=8, S=2048, D_IN=1024, D_OUT=64
#define NB   8
#define NS   2048
#define DIN  1024
#define DOUT 64

typedef __attribute__((ext_vector_type(8)))  float  v8f;
typedef __attribute__((ext_vector_type(16))) __bf16 v16bf;
typedef __attribute__((ext_vector_type(8)))  __bf16 v8bf;

static __device__ __forceinline__ unsigned short bf16_bits(float f) {
    return __builtin_bit_cast(unsigned short, static_cast<__bf16>(f));
}
static __device__ __forceinline__ v16bf cat16(v8bf lo, v8bf hi) {
    return __builtin_shufflevector(lo, hi, 0,1,2,3,4,5,6,7,8,9,10,11,12,13,14,15);
}
static __device__ __forceinline__ v16bf load_frag(const unsigned short* p) {
    v8bf lo = *reinterpret_cast<const v8bf*>(p);
    v8bf hi = *reinterpret_cast<const v8bf*>(p + 8);
    return cat16(lo, hi);
}
static __device__ __forceinline__ v8f wmma_bf16(v16bf a, v16bf b, v8f c) {
    // v_wmma_f32_16x16x32_bf16 : D = A(16x32) * B(32x16) + C(16x16 f32)
    return __builtin_amdgcn_wmma_f32_16x16x32_bf16(false, a, false, b, (short)0, c, false, false);
}

// --------------------------------------------------------------------------
// Kernel 1: swizzle W (f32 [1024x64]) -> bf16 WMMA B-fragments.
// Fragment layout index: ((mat*32 + kchunk)*4 + ntile)*512 + lane*16 + e
// element (lane,e) holds W[kchunk*32 + kpat, ntile*16 + lane%16],
// kpat = e%8 + (e/8)*16 + (lane/16)*8   (ISA 7.12.2 16-bit layout)
// --------------------------------------------------------------------------
__global__ void __launch_bounds__(256)
swizzle_w_kernel(const float* __restrict__ Wq, const float* __restrict__ Wk,
                 const float* __restrict__ Wv, unsigned short* __restrict__ Wfrag) {
    unsigned idx = blockIdx.x * 256u + threadIdx.x;   // < 3*32*4*512 = 196608
    unsigned e    = idx & 15u;
    unsigned lane = (idx >> 4) & 31u;
    unsigned nt   = (idx >> 9) & 3u;
    unsigned kc   = (idx >> 11) & 31u;
    unsigned mat  = idx >> 16;
    unsigned kpat = (e & 7u) + ((e >> 3) << 4) + ((lane >> 4) << 3);
    unsigned k = kc * 32u + kpat;
    unsigned n = nt * 16u + (lane & 15u);
    const float* src = (mat == 0) ? Wq : (mat == 1) ? Wk : Wv;
    Wfrag[idx] = bf16_bits(src[k * 64u + n]);
}

// --------------------------------------------------------------------------
// Kernel 2: fused QKV projection GEMM.  One 16-row M-tile per wave.
// acc[mat][ntile] (v8f) accumulated over 32 k-steps of depth 32.
// Writes bf16 Q/K/V row-major [B*S, 64], bias fused.
// --------------------------------------------------------------------------
__global__ void __launch_bounds__(256)
qkv_gemm_kernel(const float* __restrict__ seq, const unsigned short* __restrict__ Wfrag,
                const float* __restrict__ bq, const float* __restrict__ bk,
                const float* __restrict__ bv,
                unsigned short* __restrict__ Qbf, unsigned short* __restrict__ Kbf,
                unsigned short* __restrict__ Vbf) {
    const int lane  = threadIdx.x & 31;
    const int wave  = threadIdx.x >> 5;
    const int mtile = blockIdx.x * 8 + wave;    // 0..1023
    const int row0  = mtile * 16;
    const int hi    = lane >> 4;
    const int col   = lane & 15;
    const int row   = row0 + col;               // A-fragment row for this lane

    const v8f vzero = {};
    v8f acc[3][4];
#pragma unroll
    for (int m = 0; m < 3; ++m)
#pragma unroll
        for (int n = 0; n < 4; ++n) acc[m][n] = vzero;

    for (int kc = 0; kc < 32; ++kc) {
        const float* ar = seq + (size_t)row * DIN + kc * 32 + hi * 8;
        __builtin_prefetch(ar + 32, 0, 0);                 // global_prefetch next k-step
        float4 r0 = *(const float4*)(ar + 0);
        float4 r1 = *(const float4*)(ar + 4);
        float4 r2 = *(const float4*)(ar + 16);
        float4 r3 = *(const float4*)(ar + 20);
        v16bf a;
        a[0]=(__bf16)r0.x; a[1]=(__bf16)r0.y; a[2]=(__bf16)r0.z; a[3]=(__bf16)r0.w;
        a[4]=(__bf16)r1.x; a[5]=(__bf16)r1.y; a[6]=(__bf16)r1.z; a[7]=(__bf16)r1.w;
        a[8]=(__bf16)r2.x; a[9]=(__bf16)r2.y; a[10]=(__bf16)r2.z; a[11]=(__bf16)r2.w;
        a[12]=(__bf16)r3.x; a[13]=(__bf16)r3.y; a[14]=(__bf16)r3.z; a[15]=(__bf16)r3.w;
#pragma unroll
        for (int m = 0; m < 3; ++m) {
#pragma unroll
            for (int n = 0; n < 4; ++n) {
                const unsigned short* bp =
                    Wfrag + (((size_t)(m * 32 + kc) * 4 + n) << 9) + lane * 16;
                acc[m][n] = wmma_bf16(a, load_frag(bp), acc[m][n]);
            }
        }
    }

    const float* biases[3] = {bq, bk, bv};
    unsigned short* outs[3] = {Qbf, Kbf, Vbf};
#pragma unroll
    for (int m = 0; m < 3; ++m) {
#pragma unroll
        for (int n = 0; n < 4; ++n) {
            float bb = biases[m][n * 16 + col];
#pragma unroll
            for (int r = 0; r < 8; ++r) {
                int orow = row0 + r + 8 * hi;   // C-layout: M = r + 8*hi
                outs[m][(size_t)orow * DOUT + n * 16 + col] = bf16_bits(acc[m][n][r] + bb);
            }
        }
    }
}

// --------------------------------------------------------------------------
// Kernel 3: swizzle K,V (bf16 row-major) into WMMA B-fragments.
// Kfrag: scores B (contraction = feature): ((b*128+ktile)*2+fchunk)*512+lane*16+e
// Vfrag: context B (contraction = key):   ((b*64+kchunk)*4+ntile)*512+lane*16+e
// --------------------------------------------------------------------------
__global__ void __launch_bounds__(256)
swizzle_kv_kernel(const unsigned short* __restrict__ Kbf, const unsigned short* __restrict__ Vbf,
                  unsigned short* __restrict__ Kfrag, unsigned short* __restrict__ Vfrag) {
    unsigned idx = blockIdx.x * 256u + threadIdx.x;   // < 8*2048*64 = 1048576
    unsigned e    = idx & 15u;
    unsigned lane = (idx >> 4) & 31u;
    unsigned kpat = (e & 7u) + ((e >> 3) << 4) + ((lane >> 4) << 3);
    {   // K fragment
        unsigned fc = (idx >> 9) & 1u, kt = (idx >> 10) & 127u, b = idx >> 17;
        unsigned f = fc * 32u + kpat;
        unsigned key = kt * 16u + (lane & 15u);
        Kfrag[idx] = Kbf[((size_t)b * NS + key) * DOUT + f];
    }
    {   // V fragment
        unsigned nt = (idx >> 9) & 3u, kc = (idx >> 11) & 63u, b = idx >> 17;
        unsigned key = kc * 32u + kpat;
        unsigned f = nt * 16u + (lane & 15u);
        Vfrag[idx] = Vbf[((size_t)b * NS + key) * DOUT + f];
    }
}

// --------------------------------------------------------------------------
// Kernel 4: flash attention.  One 16-query tile per wave, 64 key-blocks of 32.
// Per block: 4 score WMMAs, mask+online softmax (shfl_xor row reductions over
// the 16-lane C-layout groups), P->bf16 A-fragment via per-wave LDS transpose,
// 4 context WMMAs with on-the-fly accumulator rescale.
// --------------------------------------------------------------------------
__global__ void __launch_bounds__(256)
attention_kernel(const unsigned short* __restrict__ Qbf,
                 const unsigned short* __restrict__ Kfrag,
                 const unsigned short* __restrict__ Vfrag,
                 const unsigned char* __restrict__ mask, float* __restrict__ out) {
    __shared__ __attribute__((aligned(64))) unsigned short pbuf[8][512]; // 16x32 bf16 per wave
    const int lane = threadIdx.x & 31;
    const int wave = threadIdx.x >> 5;
    const int g    = blockIdx.x * 8 + wave;   // 0..1023
    const int b    = g >> 7;
    const int q0   = (g & 127) * 16;
    const int hi   = lane >> 4;
    const int col  = lane & 15;

    // Q as A-fragments (two 16x32 chunks over feature dim)
    v16bf aQ[2];
    {
        const unsigned short* qp = Qbf + ((size_t)(b * NS + q0 + col)) * DOUT;
#pragma unroll
        for (int fc = 0; fc < 2; ++fc) {
            v8bf lo = *(const v8bf*)(qp + fc * 32 + hi * 8);
            v8bf hh = *(const v8bf*)(qp + fc * 32 + 16 + hi * 8);
            aQ[fc] = cat16(lo, hh);
        }
    }

    const v8f vzero = {};
    float m_run[8], l_run[8];
    v8f accC[4];
#pragma unroll
    for (int r = 0; r < 8; ++r) { m_run[r] = -3.0e38f; l_run[r] = 0.f; }
#pragma unroll
    for (int n = 0; n < 4; ++n) accC[n] = vzero;

    const float rscale = 0.022097086912079608f;  // 1/sqrt(S) per reference

    for (int kb = 0; kb < 64; ++kb) {
        float p[2][8];
#pragma unroll
        for (int h = 0; h < 2; ++h) {
            const int ktile = kb * 2 + h;
            const unsigned short* kp =
                Kfrag + ((size_t)(b * 128 + ktile) * 2) * 512 + lane * 16;
            v8f s = vzero;
            s = wmma_bf16(aQ[0], load_frag(kp), s);
            s = wmma_bf16(aQ[1], load_frag(kp + 512), s);
            const bool msk = mask[b * NS + ktile * 16 + col] != 0;   // column-wise mask
#pragma unroll
            for (int r = 0; r < 8; ++r) p[h][r] = msk ? -1.0e9f : s[r] * rscale;
        }
        // row max / correction (rows replicate across the 16-lane half-groups)
        float corr[8];
#pragma unroll
        for (int r = 0; r < 8; ++r) {
            float t = fmaxf(p[0][r], p[1][r]);
#pragma unroll
            for (int o = 1; o < 16; o <<= 1) t = fmaxf(t, __shfl_xor(t, o, 32));
            float mn = fmaxf(m_run[r], t);
            corr[r] = __expf(m_run[r] - mn);
            m_run[r] = mn;
        }
#pragma unroll
        for (int h = 0; h < 2; ++h)
#pragma unroll
            for (int r = 0; r < 8; ++r) p[h][r] = __expf(p[h][r] - m_run[r]);
#pragma unroll
        for (int r = 0; r < 8; ++r) {
            float rs = p[0][r] + p[1][r];
#pragma unroll
            for (int o = 1; o < 16; o <<= 1) rs += __shfl_xor(rs, o, 32);
            l_run[r] = l_run[r] * corr[r] + rs;
        }
#pragma unroll
        for (int n = 0; n < 4; ++n)
#pragma unroll
            for (int r = 0; r < 8; ++r) accC[n][r] *= corr[r];

        // P (C-layout) -> LDS row-major 16x32 bf16, then re-read as A-fragment
#pragma unroll
        for (int h = 0; h < 2; ++h)
#pragma unroll
            for (int r = 0; r < 8; ++r)
                pbuf[wave][(r + 8 * hi) * 32 + h * 16 + col] = bf16_bits(p[h][r]);
        asm volatile("s_wait_dscnt 0" ::: "memory");   // stores visible before cross-lane reads
        v16bf aP;
        {
            const unsigned short* pp = &pbuf[wave][col * 32];
            v8bf lo = *(const v8bf*)(pp + hi * 8);
            v8bf hh = *(const v8bf*)(pp + 16 + hi * 8);
            aP = cat16(lo, hh);
        }
        const unsigned short* vp = Vfrag + ((size_t)(b * 64 + kb) * 4) * 512 + lane * 16;
#pragma unroll
        for (int n = 0; n < 4; ++n)
            accC[n] = wmma_bf16(aP, load_frag(vp + n * 512), accC[n]);
    }

#pragma unroll
    for (int n = 0; n < 4; ++n)
#pragma unroll
        for (int r = 0; r < 8; ++r) {
            int orow = q0 + r + 8 * hi;
            float denom = fmaxf(l_run[r], 1e-30f);
            out[((size_t)(b * NS + orow)) * DOUT + n * 16 + col] = accC[n][r] / denom;
        }
}

// --------------------------------------------------------------------------
extern "C" void kernel_launch(void* const* d_in, const int* in_sizes, int n_in,
                              void* d_out, int out_size, void* d_ws, size_t ws_size,
                              hipStream_t stream) {
    const float*         seq  = (const float*)d_in[0];
    const unsigned char* mask = (const unsigned char*)d_in[1];  // jnp bool = 1 byte
    const float*         Wq   = (const float*)d_in[2];
    const float*         bq   = (const float*)d_in[3];
    const float*         Wk   = (const float*)d_in[4];
    const float*         bk   = (const float*)d_in[5];
    const float*         Wv   = (const float*)d_in[6];
    const float*         bv   = (const float*)d_in[7];
    float*               out  = (float*)d_out;

    char* ws = (char*)d_ws;
    const size_t SZ_QKV = (size_t)NB * NS * DOUT * 2;   // 2 MB each (bf16)
    unsigned short* Wfrag = (unsigned short*)(ws);                     // 384 KB used
    unsigned short* Qbf   = (unsigned short*)(ws + (512u << 10));
    unsigned short* Kbf   = (unsigned short*)(ws + (512u << 10) + 1 * SZ_QKV);
    unsigned short* Vbf   = (unsigned short*)(ws + (512u << 10) + 2 * SZ_QKV);
    unsigned short* Kfrag = (unsigned short*)(ws + (512u << 10) + 3 * SZ_QKV);
    unsigned short* Vfrag = (unsigned short*)(ws + (512u << 10) + 4 * SZ_QKV);

    (void)in_sizes; (void)n_in; (void)out_size; (void)ws_size;

    swizzle_w_kernel<<<768, 256, 0, stream>>>(Wq, Wk, Wv, Wfrag);
    qkv_gemm_kernel<<<128, 256, 0, stream>>>(seq, Wfrag, bq, bk, bv, Qbf, Kbf, Vbf);
    swizzle_kv_kernel<<<4096, 256, 0, stream>>>(Kbf, Vbf, Kfrag, Vfrag);
    attention_kernel<<<128, 256, 0, stream>>>(Qbf, Kfrag, Vfrag, mask, out);
}
